// BertAttention_33586644255286
// MI455X (gfx1250) — compile-verified, matrix operand order checked by reading, not run on hardware
//
#include <hip/hip_runtime.h>
#include <stdint.h>

typedef unsigned short bf16u;
typedef __attribute__((ext_vector_type(16))) __bf16 v16bf;
typedef __attribute__((ext_vector_type(8)))  float  v8f;
typedef __attribute__((ext_vector_type(4)))  int    v4i;

#define D_MODEL 1024
#define SEQ     2048
#define NHEAD   16
#define HDIM    64
#define MTOT    4096   // B*S

#define AS1 __attribute__((address_space(1)))
#define AS3 __attribute__((address_space(3)))

#if defined(__has_builtin)
#if __has_builtin(__builtin_amdgcn_global_load_async_to_lds_b128)
#define HAVE_ASYNC_LDS 1
#endif
#endif
#ifndef HAVE_ASYNC_LDS
#define HAVE_ASYNC_LDS 0
#endif

// ---------- helpers ----------

__device__ __forceinline__ bf16u f2bf(float f) {
  unsigned u = __float_as_uint(f);
  unsigned r = ((u >> 16) & 1u) + 0x7fffu;   // round-to-nearest-even
  return (bf16u)((u + r) >> 16);
}

struct U4x2 { uint4 a, b; };

// Load a 16x32 bf16 A/B fragment row for lane (l, hi): two contiguous 16B
// chunks per the CDNA5 16-bit matrix VGPR layout (K pairs hi*8.. and 16+hi*8..)
__device__ __forceinline__ v16bf load_frag(const bf16u* rowp, int hi) {
  U4x2 t;
  t.a = *(const uint4*)(rowp + hi * 8);
  t.b = *(const uint4*)(rowp + 16 + hi * 8);
  return __builtin_bit_cast(v16bf, t);
}

__device__ __forceinline__ v8f wmma_bf16(v16bf a, v16bf b, v8f c) {
  return __builtin_amdgcn_wmma_f32_16x16x32_bf16(
      false, a, false, b, (short)0, c, false, false);
}

// 16-byte global->LDS copy: async (ASYNCcnt-tracked, bypasses VGPRs) if available
__device__ __forceinline__ void copy16(const bf16u* g, bf16u* l) {
#if HAVE_ASYNC_LDS
  __builtin_amdgcn_global_load_async_to_lds_b128((AS1 v4i*)g, (AS3 v4i*)l, 0, 0);
#else
  *(uint4*)l = *(const uint4*)g;
#endif
}

__device__ __forceinline__ void async_wait0() {
#if HAVE_ASYNC_LDS
#if __has_builtin(__builtin_amdgcn_s_wait_asynccnt)
  __builtin_amdgcn_s_wait_asynccnt(0);
#else
  asm volatile("s_wait_asynccnt 0" ::: "memory");
#endif
#endif
}

// 16-lane row reductions via DPP16 (pure VALU, co-executes with WMMA):
// quad_perm xor1, quad_perm xor2, row_half_mirror (8), row_mirror (16)
__device__ __forceinline__ float row16_max(float x) {
  int v;
  v = __builtin_amdgcn_update_dpp(0, __float_as_int(x), 0xB1,  0xf, 0xf, true);
  x = fmaxf(x, __int_as_float(v));
  v = __builtin_amdgcn_update_dpp(0, __float_as_int(x), 0x4E,  0xf, 0xf, true);
  x = fmaxf(x, __int_as_float(v));
  v = __builtin_amdgcn_update_dpp(0, __float_as_int(x), 0x141, 0xf, 0xf, true);
  x = fmaxf(x, __int_as_float(v));
  v = __builtin_amdgcn_update_dpp(0, __float_as_int(x), 0x140, 0xf, 0xf, true);
  x = fmaxf(x, __int_as_float(v));
  return x;
}

__device__ __forceinline__ float row16_sum(float x) {
  int v;
  v = __builtin_amdgcn_update_dpp(0, __float_as_int(x), 0xB1,  0xf, 0xf, true);
  x += __int_as_float(v);
  v = __builtin_amdgcn_update_dpp(0, __float_as_int(x), 0x4E,  0xf, 0xf, true);
  x += __int_as_float(v);
  v = __builtin_amdgcn_update_dpp(0, __float_as_int(x), 0x141, 0xf, 0xf, true);
  x += __int_as_float(v);
  v = __builtin_amdgcn_update_dpp(0, __float_as_int(x), 0x140, 0xf, 0xf, true);
  x += __int_as_float(v);
  return x;
}

// ---------- conversion kernels ----------

__global__ void cvt_x_kernel(const float* __restrict__ X, bf16u* __restrict__ Xb) {
  int i = blockIdx.x * blockDim.x + threadIdx.x;   // one float4 each
  float4 v = ((const float4*)X)[i];
  union { bf16u h[4]; uint2 u; } o;
  o.h[0] = f2bf(v.x); o.h[1] = f2bf(v.y); o.h[2] = f2bf(v.z); o.h[3] = f2bf(v.w);
  ((uint2*)Xb)[i] = o.u;
}

// WT[z][n][k] = W_z[k][n]  (bf16, transposed so B-fragments load row-wise)
__global__ void cvt_wT_kernel(const float* __restrict__ W0, const float* __restrict__ W1,
                              const float* __restrict__ W2, const float* __restrict__ W3,
                              bf16u* __restrict__ WT) {
  int z = blockIdx.y;
  const float* W = (z == 0) ? W0 : (z == 1) ? W1 : (z == 2) ? W2 : W3;
  int idx = blockIdx.x * 256 + threadIdx.x;        // 0 .. 1M-1
  int n = idx >> 10, k = idx & 1023;
  WT[(size_t)z * 1024 * 1024 + idx] = f2bf(W[k * 1024 + n]);
}

// ---------- tiled bf16 WMMA GEMM: C = A(4096x1024) * WT^T + bias ----------
// Double-buffered LDS staging with async global->LDS copies.
// MODE 0: QKV projection, writes bf16 into [B,H,S,Dh], z = blockIdx.z selects matrix
// MODE 1: output projection, adds residual, writes fp32 Hout
template <int MODE>
__global__ void gemm_kernel(const bf16u* __restrict__ A,
                            const bf16u* __restrict__ WT,
                            const float* __restrict__ bias0,
                            const float* __restrict__ bias1,
                            const float* __restrict__ bias2,
                            bf16u* __restrict__ qkv,
                            const float* __restrict__ resid,
                            float* __restrict__ Hout) {
  __shared__ bf16u As[2][128 * 40];  // 128x32 tile, stride 40 (80B, 16B-aligned)
  __shared__ bf16u Bs[2][128 * 40];

  const int tid  = threadIdx.x;
  const int wave = tid >> 5, lane = tid & 31;
  const int l = lane & 15, hi = lane >> 4;
  const int wm = wave >> 2, wn = wave & 3;          // 2x4 wave grid
  const int z = (MODE == 0) ? blockIdx.z : 0;
  const int mbase = blockIdx.y * 128;
  const int nbase = blockIdx.x * 128;
  const bf16u* Bt = WT + (size_t)z * (1024 * 1024);
  const float* bias = (MODE == 0) ? ((z == 0) ? bias0 : (z == 1) ? bias1 : bias2)
                                  : bias0;

  v8f zinit = {};
  v8f acc[4][2];
#pragma unroll
  for (int i = 0; i < 4; i++)
#pragma unroll
    for (int j = 0; j < 2; j++) acc[i][j] = zinit;

  const int r0 = tid >> 2, q = tid & 3;             // 4 threads/row, 16B each
  const bf16u* ag = A  + (size_t)(mbase + r0) * 1024 + q * 8;
  const bf16u* bg = Bt + (size_t)(nbase + r0) * 1024 + q * 8;

  auto stage = [&](int buf, int kb) {
    const int ko = kb * 32;
    copy16(ag + ko,                      &As[buf][r0 * 40 + q * 8]);
    copy16(ag + ko + (size_t)64 * 1024,  &As[buf][(r0 + 64) * 40 + q * 8]);
    copy16(bg + ko,                      &Bs[buf][r0 * 40 + q * 8]);
    copy16(bg + ko + (size_t)64 * 1024,  &Bs[buf][(r0 + 64) * 40 + q * 8]);
  };

  stage(0, 0);
  async_wait0();
  __syncthreads();

  int buf = 0;
  for (int kb = 0; kb < 32; ++kb) {
    if (kb + 1 < 32) stage(buf ^ 1, kb + 1);   // prefetch next tile

    v16bf bfrag[2];
#pragma unroll
    for (int j = 0; j < 2; j++)
      bfrag[j] = load_frag(&Bs[buf][(wn * 32 + j * 16 + l) * 40], hi);
#pragma unroll
    for (int i = 0; i < 4; i++) {
      v16bf afrag = load_frag(&As[buf][(wm * 64 + i * 16 + l) * 40], hi);
#pragma unroll
      for (int j = 0; j < 2; j++)
        acc[i][j] = wmma_bf16(afrag, bfrag[j], acc[i][j]);
    }

    async_wait0();
    __syncthreads();
    buf ^= 1;
  }

  // epilogue: C-layout value (lane l, half hi, comp r) = C[r + hi*8][l] of a 16x16 tile
#pragma unroll
  for (int i = 0; i < 4; i++) {
#pragma unroll
    for (int j = 0; j < 2; j++) {
      const int n = nbase + wn * 32 + j * 16 + l;
      const float bn = bias[n];
#pragma unroll
      for (int r = 0; r < 8; r++) {
        const int m = mbase + wm * 64 + i * 16 + r + hi * 8;
        float val = acc[i][j][r] + bn;
        if (MODE == 0) {
          int bb = m >> 11, s = m & 2047;
          int h = n >> 6, dh = n & 63;
          qkv[(size_t)z * MTOT * D_MODEL +
              (((size_t)(bb * NHEAD + h) * SEQ + s) * HDIM + dh)] = f2bf(val);
        } else {
          Hout[(size_t)m * D_MODEL + n] = val + resid[(size_t)m * D_MODEL + n];
        }
      }
    }
  }
}

// ---------- flash-style attention, one (b,h) x 128 query rows per block ----------

__global__ void attn_kernel(const bf16u* __restrict__ qkv, bf16u* __restrict__ ctx) {
  __shared__ bf16u Qs[128 * 72];      // 128 q-rows x 64, stride 72 (144B)
  __shared__ bf16u Ks[64 * 72];       // 64 k-rows x 64
  __shared__ bf16u Vts[64 * 72];      // V transposed: [d][t]
  __shared__ bf16u Ps[8 * 16 * 72];   // per-wave P tile, 16 x 64

  const int tid  = threadIdx.x;
  const int wave = tid >> 5, lane = tid & 31;
  const int l = lane & 15, hi = lane >> 4;
  const int bh = blockIdx.y;
  const int qbase = blockIdx.x * 128;
  const size_t headoff = (size_t)bh * SEQ * HDIM;
  const bf16u* Qp = qkv + headoff;
  const bf16u* Kp = qkv + (size_t)MTOT * D_MODEL + headoff;
  const bf16u* Vp = qkv + (size_t)2 * MTOT * D_MODEL + headoff;

  // stage Q tile (16KB) async: 2 threads/row, 64B each
  {
    int row = tid >> 1, half = tid & 1;
    const bf16u* g = Qp + (size_t)(qbase + row) * HDIM + half * 32;
    bf16u* d = &Qs[row * 72 + half * 32];
#pragma unroll
    for (int c = 0; c < 4; c++) copy16(g + c * 8, d + c * 8);
  }

  v8f zinit = {};
  v8f sacc[4], o[4];
  float mrun[8], lrun[8];
#pragma unroll
  for (int n = 0; n < 4; n++) o[n] = zinit;
#pragma unroll
  for (int r = 0; r < 8; r++) { mrun[r] = -1e30f; lrun[r] = 0.f; }

  for (int kb = 0; kb < SEQ / 64; ++kb) {
    const int kt0 = kb * 64;
    __syncthreads();   // previous iteration's reads of Ks/Vts complete
    // stage K tile async: 4 threads/row, 32B each
    {
      int row = tid >> 2, q = tid & 3;
      const bf16u* g = Kp + (size_t)(kt0 + row) * HDIM + q * 16;
      bf16u* d = &Ks[row * 72 + q * 16];
      copy16(g, d);
      copy16(g + 8, d + 8);
    }
    // stage V transposed: coalesced global read, scattered LDS b16 writes
#pragma unroll
    for (int c = 0; c < 2; c++) {
      int idx = tid + c * 256;
      int t = idx >> 3, dc = (idx & 7) * 8;
      union { uint4 u; bf16u h[8]; } tmp;
      tmp.u = *(const uint4*)(Vp + (size_t)(kt0 + t) * HDIM + dc);
#pragma unroll
      for (int e = 0; e < 8; e++) Vts[(dc + e) * 72 + t] = tmp.h[e];
    }
    async_wait0();
    __syncthreads();

    // scores: S(16x64) = Q(16x64) * K^T   (K row-major == Bt layout)
#pragma unroll
    for (int j = 0; j < 4; j++) sacc[j] = zinit;
#pragma unroll
    for (int ks = 0; ks < 2; ks++) {
      v16bf afrag = load_frag(&Qs[(wave * 16 + l) * 72 + ks * 32], hi);
#pragma unroll
      for (int j = 0; j < 4; j++) {
        v16bf bfrag = load_frag(&Ks[(j * 16 + l) * 72 + ks * 32], hi);
        sacc[j] = wmma_bf16(afrag, bfrag, sacc[j]);
      }
    }

    // online softmax; 16-lane row reductions via DPP (no LDS pipe)
#pragma unroll
    for (int r = 0; r < 8; r++) {
      float m_b = -1e30f;
#pragma unroll
      for (int j = 0; j < 4; j++) {
        float t = sacc[j][r] * 0.125f;   // 1/sqrt(64)
        sacc[j][r] = t;
        m_b = fmaxf(m_b, t);
      }
      m_b = row16_max(m_b);
      float m_new = fmaxf(mrun[r], m_b);
      float alpha = __expf(mrun[r] - m_new);
      mrun[r] = m_new;
      float srow = 0.f;
#pragma unroll
      for (int j = 0; j < 4; j++) {
        float p = __expf(sacc[j][r] - m_new);
        sacc[j][r] = p;
        srow += p;
      }
      srow = row16_sum(srow);
      lrun[r] = lrun[r] * alpha + srow;
#pragma unroll
      for (int n = 0; n < 4; n++) o[n][r] *= alpha;
    }

    // transpose P from C-layout to A-layout through per-wave LDS region
#pragma unroll
    for (int j = 0; j < 4; j++)
#pragma unroll
      for (int r = 0; r < 8; r++)
        Ps[(wave * 16 + r + hi * 8) * 72 + j * 16 + l] = f2bf(sacc[j][r]);

    asm volatile("s_wait_dscnt 0" ::: "memory");   // wave-local LDS RAW fence

    // ctx(16x64) += P(16x64) * V(64x64)   (Vts[d][t] == Bt layout)
#pragma unroll
    for (int ks = 0; ks < 2; ks++) {
      v16bf afrag = load_frag(&Ps[(wave * 16 + l) * 72 + ks * 32], hi);
#pragma unroll
      for (int n = 0; n < 4; n++) {
        v16bf bfrag = load_frag(&Vts[(n * 16 + l) * 72 + ks * 32], hi);
        o[n] = wmma_bf16(afrag, bfrag, o[n]);
      }
    }
  }

  // normalize and store ctx back to [B*S, D] bf16
  const int bb = bh >> 4, h = bh & 15;
#pragma unroll
  for (int r = 0; r < 8; r++) {
    float inv = 1.0f / lrun[r];
    int s = qbase + wave * 16 + r + hi * 8;
    size_t mrow = ((size_t)bb * SEQ + s) * D_MODEL;
#pragma unroll
    for (int n = 0; n < 4; n++) {
      int col = h * HDIM + n * 16 + l;
      ctx[mrow + col] = f2bf(o[n][r] * inv);
    }
  }
}

// ---------- LayerNorm over D=1024 per row ----------

__global__ void ln_kernel(const float* __restrict__ Hb, const float* __restrict__ gamma,
                          const float* __restrict__ beta, float* __restrict__ out) {
  __shared__ float rs[256], rss[256];
  const int row = blockIdx.x, tid = threadIdx.x;
  const float* hp = Hb + (size_t)row * D_MODEL;
  float s = 0.f, ss = 0.f;
  for (int i = tid; i < D_MODEL; i += 256) {
    float x = hp[i]; s += x; ss += x * x;
  }
  rs[tid] = s; rss[tid] = ss;
  __syncthreads();
  for (int st = 128; st > 0; st >>= 1) {
    if (tid < st) { rs[tid] += rs[tid + st]; rss[tid] += rss[tid + st]; }
    __syncthreads();
  }
  float mu   = rs[0] * (1.0f / D_MODEL);
  float var  = rss[0] * (1.0f / D_MODEL) - mu * mu;
  float rstd = rsqrtf(var + 1e-12f);
  for (int i = tid; i < D_MODEL; i += 256) {
    out[(size_t)row * D_MODEL + i] = (hp[i] - mu) * rstd * gamma[i] + beta[i];
  }
}

// ---------- launch ----------

extern "C" void kernel_launch(void* const* d_in, const int* in_sizes, int n_in,
                              void* d_out, int out_size, void* d_ws, size_t ws_size,
                              hipStream_t stream) {
  (void)in_sizes; (void)n_in; (void)out_size; (void)ws_size;
  const float* X     = (const float*)d_in[0];
  const float* Wq    = (const float*)d_in[1];
  const float* bq    = (const float*)d_in[2];
  const float* Wk    = (const float*)d_in[3];
  const float* bk    = (const float*)d_in[4];
  const float* Wv    = (const float*)d_in[5];
  const float* bv    = (const float*)d_in[6];
  const float* Wo    = (const float*)d_in[7];
  const float* bo    = (const float*)d_in[8];
  const float* gamma = (const float*)d_in[9];
  const float* beta  = (const float*)d_in[10];
  float* out = (float*)d_out;

  char* ws = (char*)d_ws;
  bf16u* Xb  = (bf16u*)ws;                              //  8 MB: X in bf16
  bf16u* WT  = (bf16u*)(ws + 8ull  * 1024 * 1024);      //  8 MB: 4 transposed weights
  bf16u* QKV = (bf16u*)(ws + 16ull * 1024 * 1024);      // 24 MB: Q,K,V [B,H,S,Dh]
  bf16u* Ctx = (bf16u*)(ws + 40ull * 1024 * 1024);      //  8 MB: context [B*S, D]
  float* Hb  = (float*)(ws + 48ull * 1024 * 1024);      // 16 MB: pre-LN fp32

  cvt_x_kernel <<<dim3(MTOT * D_MODEL / 1024), 256, 0, stream>>>(X, Xb);
  cvt_wT_kernel<<<dim3(4096, 4),               256, 0, stream>>>(Wq, Wk, Wv, Wo, WT);
  gemm_kernel<0><<<dim3(8, 32, 3), 256, 0, stream>>>(Xb, WT, bq, bk, bv, QKV,
                                                     nullptr, nullptr);
  attn_kernel  <<<dim3(16, 32),    256, 0, stream>>>(QKV, Ctx);
  gemm_kernel<1><<<dim3(8, 32, 1), 256, 0, stream>>>(Ctx, WT + 3ull * 1024 * 1024,
                                                     bo, nullptr, nullptr, nullptr,
                                                     X, Hb);
  ln_kernel    <<<dim3(4096),      256, 0, stream>>>(Hb, gamma, beta, out);
}
